// MSSD_72507637891495
// MI455X (gfx1250) — matrix operand order repaired and checked by reading, not compiled, not run on hardware
//
#include <hip/hip_runtime.h>
#include <hip/hip_bf16.h>
#include <math.h>

typedef __attribute__((ext_vector_type(16))) _Float16 v16h;
typedef __attribute__((ext_vector_type(8)))  _Float16 v8h;
typedef __attribute__((ext_vector_type(8)))  float    v8f;

#define SHUF16(lo, hi) __builtin_shufflevector(lo, hi, 0,1,2,3,4,5,6,7,8,9,10,11,12,13,14,15)

// ---------------------------------------------------------------------------
// Zero-fill guard buffer (used to redirect invalid halo / edge-channel loads)
// ---------------------------------------------------------------------------
__global__ __launch_bounds__(256) void vgg_zerof(float* __restrict__ p, int n) {
  int i = blockIdx.x * 256 + threadIdx.x;
  if (i < n) p[i] = 0.f;
}

// ---------------------------------------------------------------------------
// Weight packing: fp32 OIHW (Cout,Cin,3,3) -> fp16 [Cout][Kpad], K = (ky*3+kx)*Cin + c
// ---------------------------------------------------------------------------
__global__ __launch_bounds__(256) void vgg_pack_w(const float* __restrict__ w,
                                                  _Float16* __restrict__ w16,
                                                  int Cin, int Cout, int Kpad) {
  long i = (long)blockIdx.x * 256 + threadIdx.x;
  long total = (long)Cout * Kpad;
  if (i >= total) return;
  int n = (int)(i / Kpad);
  int k = (int)(i - (long)n * Kpad);
  int kidx = k / Cin;
  int c = k - kidx * Cin;
  float v = 0.f;
  if (kidx < 9) {
    int ky = kidx / 3, kx = kidx - ky * 3;
    v = w[(((long)n * Cin + c) * 3 + ky) * 3 + kx];
  }
  w16[i] = (_Float16)v;
}

// Fold conv bias + BN (eval) into per-channel alpha/beta
__global__ __launch_bounds__(256) void vgg_prep_affine(
    const float* __restrict__ b, const float* __restrict__ gamma,
    const float* __restrict__ beta, const float* __restrict__ mean,
    const float* __restrict__ var, float* __restrict__ alpha,
    float* __restrict__ betaO, int C) {
  int i = blockIdx.x * 256 + threadIdx.x;
  if (i >= C) return;
  float s = gamma[i] * rsqrtf(var[i] + 1e-5f);
  alpha[i] = s;
  betaO[i] = (b[i] - mean[i]) * s + beta[i];
}

// NCHW fp32 -> NHWC fp16
__global__ __launch_bounds__(256) void vgg_nchw_to_nhwc(
    const float* __restrict__ in, _Float16* __restrict__ out,
    int B, int C, int H, int W) {
  long i = (long)blockIdx.x * 256 + threadIdx.x;
  long total = (long)B * C * H * W;
  if (i >= total) return;
  int c = (int)(i % C);
  long t = i / C;
  int x = (int)(t % W); t /= W;
  int y = (int)(t % H);
  int bb = (int)(t / H);
  out[i] = (_Float16)in[(((long)bb * C + c) * H + y) * W + x];
}

// 2x2/2 maxpool, NHWC fp16. H,W are INPUT dims.
__global__ __launch_bounds__(256) void vgg_maxpool2(
    const _Float16* __restrict__ in, _Float16* __restrict__ out,
    int B, int C, int H, int W) {
  int Ho = H >> 1, Wo = W >> 1;
  long i = (long)blockIdx.x * 256 + threadIdx.x;
  long total = (long)B * Ho * Wo * C;
  if (i >= total) return;
  int c = (int)(i % C);
  long t = i / C;
  int x = (int)(t % Wo); t /= Wo;
  int y = (int)(t % Ho);
  int bb = (int)(t / Ho);
  long base = (((long)(bb * H + 2 * y) * W) + 2 * x) * C + c;
  float v0 = (float)in[base];
  float v1 = (float)in[base + C];
  float v2 = (float)in[base + (long)W * C];
  float v3 = (float)in[base + (long)W * C + C];
  out[i] = (_Float16)fmaxf(fmaxf(v0, v1), fmaxf(v2, v3));
}

// ---------------------------------------------------------------------------
// FAST implicit-GEMM conv (requires Cin % 32 == 0).
// One wave computes a 32(M pixels) x 64(N channels) tile: 2x4 register tiling,
// 8 fp32 accumulators, 8 v_wmma_f32_16x16x32_f16 per 32-channel chunk from
// 12 unconditional b128 loads (invalid halo/edge rows are redirected into a
// zeroed guard buffer, so the inner loop has NO exec-mask manipulation).
// Fragment element->K map used here, same for A and B: K=half*16+e (one contiguous
// 16-half run, loaded as two b128 pieces).
// ---------------------------------------------------------------------------
__global__ __launch_bounds__(256) void vgg_conv_wmma64(
    const _Float16* __restrict__ act, const _Float16* __restrict__ wgt,
    const _Float16* __restrict__ zbuf,
    const float* __restrict__ alpha, const float* __restrict__ beta,
    _Float16* __restrict__ out, float* __restrict__ headOut,
    int B, int H, int W, int Cin, int Cout, int Kpad,
    int mode, int headOff, int headStride) {
  const int lane = threadIdx.x & 31;
  const int wave = threadIdx.x >> 5;
  const int HW = H * W;
  const long Mtot = (long)B * HW;
  const int tilesN = (Cout + 63) >> 6;
  const long tilesM = (Mtot + 31) >> 5;
  long tileIdx = (long)blockIdx.x * 8 + wave;        // wave-uniform
  if (tileIdx >= tilesM * tilesN) return;            // wave-uniform exit
  const int tileN = (int)(tileIdx % tilesN);
  const long tileM = tileIdx / tilesN;
  const int row = lane & 15, half = lane >> 4;

  // Two A pixel rows per lane (M sub-tiles)
  int px[2], py[2], pb[2];
  bool pv[2];
#pragma unroll
  for (int mi = 0; mi < 2; ++mi) {
    long m = tileM * 32 + mi * 16 + row;
    pv[mi] = m < Mtot;
    long mc = pv[mi] ? m : 0;
    int bb = (int)(mc / HW);
    int rem = (int)(mc - (long)bb * HW);
    pb[mi] = bb;
    py[mi] = rem / W;
    px[mi] = rem - (rem / W) * W;
  }

  // Four B weight rows per lane (N sub-tiles); invalid rows -> zero buffer
  int nn[4];
  bool nv[4];
  const _Float16* wr[4];
#pragma unroll
  for (int ni = 0; ni < 4; ++ni) {
    nn[ni] = tileN * 64 + ni * 16 + row;
    nv[ni] = nn[ni] < Cout;
    wr[ni] = nv[ni] ? (wgt + (long)nn[ni] * Kpad + half * 16) : (zbuf + half * 16);
  }

  v8f acc[2][4];
#pragma unroll
  for (int mi = 0; mi < 2; ++mi)
#pragma unroll
    for (int ni = 0; ni < 4; ++ni) acc[mi][ni] = (v8f){};

  int ky = 0, kx = 0;
  for (int kidx = 0; kidx < 9; ++kidx) {
    // Per-tap A row pointers; invalid halo rows -> zero buffer
    const _Float16* p[2];
#pragma unroll
    for (int mi = 0; mi < 2; ++mi) {
      int yy = py[mi] + ky - 1, xx = px[mi] + kx - 1;
      bool v = pv[mi] && yy >= 0 && yy < H && xx >= 0 && xx < W;
      p[mi] = v ? (act + ((long)(pb[mi] * H + yy) * W + xx) * Cin + half * 16)
                : (zbuf + half * 16);
    }
    const long wofs = (long)kidx * Cin;
#pragma unroll
    for (int ni = 0; ni < 4; ++ni)
      __builtin_prefetch(wr[ni] + wofs + Cin, 0, 0);  // next tap's weights

    for (int c0 = 0; c0 < Cin; c0 += 32) {
      v16h a[2], f[4];
#pragma unroll
      for (int mi = 0; mi < 2; ++mi) {
        v8h lo = *(const v8h*)(p[mi] + c0);
        v8h hi = *(const v8h*)(p[mi] + c0 + 8);
        a[mi] = SHUF16(lo, hi);
      }
#pragma unroll
      for (int ni = 0; ni < 4; ++ni) {
        v8h lo = *(const v8h*)(wr[ni] + wofs + c0);
        v8h hi = *(const v8h*)(wr[ni] + wofs + c0 + 8);
        f[ni] = SHUF16(lo, hi);
      }
#pragma unroll
      for (int mi = 0; mi < 2; ++mi)
#pragma unroll
        for (int ni = 0; ni < 4; ++ni)
          acc[mi][ni] = __builtin_amdgcn_wmma_f32_16x16x32_f16(
              false, a[mi], false, f[ni], (short)0, acc[mi][ni], false, false);
    }
    if (++kx == 3) { kx = 0; ++ky; }
  }

  // Epilogue. D layout: VGPR r, lane l -> M = r + (l>>4)*8, N = l&15
#pragma unroll
  for (int ni = 0; ni < 4; ++ni) {
    float al = 0.f, be = 0.f;
    if (nv[ni]) { al = alpha[nn[ni]]; be = beta[nn[ni]]; }
#pragma unroll
    for (int mi = 0; mi < 2; ++mi) {
#pragma unroll
      for (int r = 0; r < 8; ++r) {
        long mo = tileM * 32 + mi * 16 + r + half * 8;
        if (mo < Mtot && nv[ni]) {
          float v = acc[mi][ni][r] * al + be;
          v = v > 0.f ? v : 0.f;                     // ReLU
          if (mode == 0) {
            out[mo * Cout + nn[ni]] = (_Float16)v;
          } else {
            int b2 = (int)(mo / HW);
            int pix = (int)(mo - (long)b2 * HW);
            float dv = (nn[ni] == 2 || nn[ni] == 3) ? __expf(v)
                                                    : 1.f / (1.f + __expf(-v));
            headOut[((long)b2 * 5 + nn[ni]) * headStride + headOff + pix] = dv;
          }
        }
      }
    }
  }
}

// ---------------------------------------------------------------------------
// Generic implicit-GEMM conv (any Cin; used only for layer 0, Cin=3, K=27->32)
// ---------------------------------------------------------------------------
__global__ __launch_bounds__(256) void vgg_conv_wmma(
    const _Float16* __restrict__ act, const _Float16* __restrict__ wgt,
    const float* __restrict__ alpha, const float* __restrict__ beta,
    _Float16* __restrict__ out,
    int B, int H, int W, int Cin, int Cout, int Kpad) {
  const int lane = threadIdx.x & 31;
  const int wave = threadIdx.x >> 5;
  const int HW = H * W;
  const long Mtot = (long)B * HW;
  const int tilesN = (Cout + 15) >> 4;
  const long tilesM = (Mtot + 15) >> 4;
  long tileIdx = (long)blockIdx.x * 8 + wave;
  if (tileIdx >= tilesM * tilesN) return;
  const int tileN = (int)(tileIdx % tilesN);
  const long tileM = tileIdx / tilesN;
  const int row = lane & 15, half = lane >> 4;

  long m = tileM * 16 + row;
  const bool mValid = (m < Mtot);
  long mc = mValid ? m : 0;
  int bb = (int)(mc / HW);
  int rem = (int)(mc - (long)bb * HW);
  int y = rem / W, x = rem - (rem / W) * W;
  const int nB = tileN * 16 + row;
  const bool nValid = (nB < Cout);

  v8f acc = {};
  for (int k0 = 0; k0 < Kpad; k0 += 32) {
    v16h a, bm;
#pragma unroll
    for (int e = 0; e < 16; ++e) {
      int ka = k0 + (half << 4) + e;
      int kidx = ka / Cin;
      int cc = ka - kidx * Cin;
      _Float16 av = (_Float16)0.f;
      if (mValid && kidx < 9) {
        int ky = kidx / 3, kx = kidx - ky * 3;
        int yy = y + ky - 1, xx = x + kx - 1;
        if (yy >= 0 && yy < H && xx >= 0 && xx < W)
          av = act[(((long)(bb * H + yy) * W) + xx) * Cin + cc];
      }
      a[e] = av;
      int kb = k0 + (half << 4) + e;
      bm[e] = nValid ? wgt[(long)nB * Kpad + kb] : (_Float16)0.f;
    }
    acc = __builtin_amdgcn_wmma_f32_16x16x32_f16(false, a, false, bm, (short)0, acc, false, false);
  }

  const int nO = tileN * 16 + row;
  float al = 0.f, be = 0.f;
  if (nO < Cout) { al = alpha[nO]; be = beta[nO]; }
#pragma unroll
  for (int r = 0; r < 8; ++r) {
    long mo = tileM * 16 + r + half * 8;
    if (mo < Mtot && nO < Cout) {
      float v = acc[r] * al + be;
      out[mo * Cout + nO] = (_Float16)(v > 0.f ? v : 0.f);
    }
  }
}

// ---------------------------------------------------------------------------
// Host orchestration
// ---------------------------------------------------------------------------
extern "C" void kernel_launch(void* const* d_in, const int* in_sizes, int n_in,
                              void* d_out, int out_size, void* d_ws, size_t ws_size,
                              hipStream_t stream) {
  (void)out_size; (void)ws_size;
  static const int LCIN[16]  = {3,64,  64,128, 128,256,256, 256,512,512, 512,512,512, 256,512,512};
  static const int LCOUT[16] = {64,64, 128,128, 256,256,256, 512,512,512, 512,512,512, 5,5,5};
  static const int LH[16]    = {512,512, 256,256, 128,128,128, 64,64,64, 32,32,32, 64,32,16};
  int KPAD[16];
  for (int l = 0; l < 16; ++l) KPAD[l] = ((LCIN[l] * 9 + 31) / 32) * 32;

  // ---- input pointer mapping (insertion-order vs sorted pytree order)
  const long XN = 4L * 3 * 512 * 512;
  bool xFirst = (n_in > 0 && (long)in_sizes[0] == XN);
  int pstart = xFirst ? 1 : 0;
  const float* xPtr = (const float*)d_in[xFirst ? 0 : (n_in - 1)];
  const float *Wp[16], *Bp[16], *Gp[16], *Btp[16], *Mp[16], *Vp[16];
  for (int l = 0; l < 16; ++l) {
    int base = pstart + l * 6;
    long wElems = (long)LCOUT[l] * LCIN[l] * 9;
    if ((long)in_sizes[base] == wElems) {            // w,b,gamma,beta,mean,var
      Wp[l] = (const float*)d_in[base + 0];  Bp[l]  = (const float*)d_in[base + 1];
      Gp[l] = (const float*)d_in[base + 2];  Btp[l] = (const float*)d_in[base + 3];
      Mp[l] = (const float*)d_in[base + 4];  Vp[l]  = (const float*)d_in[base + 5];
    } else {                                          // b,beta,gamma,mean,var,w
      Bp[l] = (const float*)d_in[base + 0];  Btp[l] = (const float*)d_in[base + 1];
      Gp[l] = (const float*)d_in[base + 2];  Mp[l]  = (const float*)d_in[base + 3];
      Vp[l] = (const float*)d_in[base + 4];  Wp[l]  = (const float*)d_in[base + 5];
    }
  }

  // ---- workspace carve-up
  char* ws = (char*)d_ws;
  size_t off = 0;
  auto alloc = [&](size_t bytes) { size_t o = off; off += (bytes + 255) & ~(size_t)255; return o; };
  const size_t BIG = (size_t)4 * 512 * 512 * 64 * 2;   // 128 MB max activation (fp16)
  size_t offA    = alloc(BIG);
  size_t offB    = alloc(BIG);
  size_t offIn   = alloc((size_t)4 * 512 * 512 * 3 * 2);
  size_t offP9   = alloc((size_t)4 * 64 * 64 * 256 * 2);
  size_t offP13  = alloc((size_t)4 * 32 * 32 * 512 * 2);
  size_t offP17  = alloc((size_t)4 * 16 * 16 * 512 * 2);
  size_t offZero = alloc(16384);                        // zero guard (>= Kpad*2+32)
  size_t offW[16], offAB[16];
  for (int l = 0; l < 16; ++l) {
    offW[l]  = alloc((size_t)LCOUT[l] * KPAD[l] * 2);
    offAB[l] = alloc((size_t)LCOUT[l] * 2 * 4);
  }

  // ---- zero the guard buffer (every call: deterministic)
  vgg_zerof<<<dim3(16), dim3(256), 0, stream>>>((float*)(ws + offZero), 4096);

  // ---- weight packing + BN folding
  for (int l = 0; l < 16; ++l) {
    long totW = (long)LCOUT[l] * KPAD[l];
    vgg_pack_w<<<dim3((unsigned)((totW + 255) / 256)), dim3(256), 0, stream>>>(
        Wp[l], (_Float16*)(ws + offW[l]), LCIN[l], LCOUT[l], KPAD[l]);
    vgg_prep_affine<<<dim3((unsigned)((LCOUT[l] + 255) / 256)), dim3(256), 0, stream>>>(
        Bp[l], Gp[l], Btp[l], Mp[l], Vp[l],
        (float*)(ws + offAB[l]), (float*)(ws + offAB[l]) + LCOUT[l], LCOUT[l]);
  }

  // ---- input layout conversion
  vgg_nchw_to_nhwc<<<dim3((unsigned)((XN + 255) / 256)), dim3(256), 0, stream>>>(
      xPtr, (_Float16*)(ws + offIn), 4, 3, 512, 512);

  auto conv = [&](int l, const void* src, void* dst, int mode, int headOff) {
    long Mtot = 4L * LH[l] * LH[l];
    const float* al = (const float*)(ws + offAB[l]);
    if (LCIN[l] % 32 == 0) {
      long tiles = ((Mtot + 31) / 32) * (long)((LCOUT[l] + 63) / 64);
      unsigned blocks = (unsigned)((tiles + 7) / 8);
      vgg_conv_wmma64<<<dim3(blocks), dim3(256), 0, stream>>>(
          (const _Float16*)src, (const _Float16*)(ws + offW[l]),
          (const _Float16*)(ws + offZero),
          al, al + LCOUT[l], (_Float16*)dst, (float*)d_out,
          4, LH[l], LH[l], LCIN[l], LCOUT[l], KPAD[l], mode, headOff, 5376);
    } else {
      long tiles = ((Mtot + 15) / 16) * (long)((LCOUT[l] + 15) / 16);
      unsigned blocks = (unsigned)((tiles + 7) / 8);
      vgg_conv_wmma<<<dim3(blocks), dim3(256), 0, stream>>>(
          (const _Float16*)src, (const _Float16*)(ws + offW[l]),
          al, al + LCOUT[l], (_Float16*)dst,
          4, LH[l], LH[l], LCIN[l], LCOUT[l], KPAD[l]);
    }
  };
  auto pool = [&](const void* src, void* dst, int C, int Hin) {
    long tot = 4L * (Hin / 2) * (Hin / 2) * C;
    vgg_maxpool2<<<dim3((unsigned)((tot + 255) / 256)), dim3(256), 0, stream>>>(
        (const _Float16*)src, (_Float16*)dst, 4, C, Hin, Hin);
  };

  void* A   = ws + offA;
  void* B   = ws + offB;
  void* IN  = ws + offIn;
  void* P9  = ws + offP9;
  void* P13 = ws + offP13;
  void* P17 = ws + offP17;

  // Block 1 @512
  conv(0, IN, A, 0, 0);
  conv(1, A, B, 0, 0);
  pool(B, A, 64, 512);            // -> 256
  // Block 2 @256
  conv(2, A, B, 0, 0);
  conv(3, B, A, 0, 0);
  pool(A, B, 128, 256);           // -> 128
  // Block 3 @128
  conv(4, B, A, 0, 0);
  conv(5, A, B, 0, 0);
  conv(6, B, A, 0, 0);
  pool(A, P9, 256, 128);          // -> 64  (kept for head 0)
  // Block 4 @64
  conv(7, P9, A, 0, 0);
  conv(8, A, B, 0, 0);
  conv(9, B, A, 0, 0);
  pool(A, P13, 512, 64);          // -> 32  (kept for head 1)
  // Block 5 @32
  conv(10, P13, A, 0, 0);
  conv(11, A, B, 0, 0);
  conv(12, B, A, 0, 0);
  pool(A, P17, 512, 32);          // -> 16  (kept for head 2)

  // SSD heads -> decoded fp32 directly into d_out [4,5,5376]
  conv(13, P9,  A, 1, 0);         // 64x64 -> [0,4096)
  conv(14, P13, A, 1, 4096);      // 32x32 -> [4096,5120)
  conv(15, P17, A, 1, 5120);      // 16x16 -> [5120,5376)
}
